// HPF_18640158065293
// MI455X (gfx1250) — matrix operand order repaired and checked by reading, not compiled, not run on hardware
//
#include <hip/hip_runtime.h>

// ---------------------------------------------------------------------------
// Types for CDNA5 WMMA (wave32): bf16 16x16x32, f32 accumulate.
// ---------------------------------------------------------------------------
typedef __attribute__((ext_vector_type(8)))  unsigned short u16x8;
typedef __attribute__((ext_vector_type(16))) unsigned short u16x16;
typedef __attribute__((ext_vector_type(16))) __bf16        v16bf;
typedef __attribute__((ext_vector_type(8)))  float         v8f;

#define DEVINL __device__ __forceinline__

// fp32 <-> bf16 via bit manipulation (round-to-nearest-even on pack).
DEVINL float bf2f(unsigned short u) {
    union { unsigned int i; float f; } c; c.i = ((unsigned int)u) << 16; return c.f;
}
DEVINL unsigned short f2bf(float f) {
    union { float f; unsigned int i; } c; c.f = f;
    unsigned int i = c.i;
    return (unsigned short)((i + 0x7FFFu + ((i >> 16) & 1u)) >> 16);
}

DEVINL v8f wmma_bf16(v16bf a, v16bf b, v8f c) {
    // (neg_a, A, neg_b, B, c_mod, C, reuse_a, reuse_b)
    return __builtin_amdgcn_wmma_f32_16x16x32_bf16(false, a, false, b, (short)0, c,
                                                   false, false);
}

DEVINL v16bf cat_u8(u16x8 lo, u16x8 hi) {
    u16x16 u = __builtin_shufflevector(lo, hi, 0,1,2,3,4,5,6,7,8,9,10,11,12,13,14,15);
    return __builtin_bit_cast(v16bf, u);
}

// A operand (16x32, MxK), row-major source base[row*ld + k].
// Lanes 0-15 hold M=lane, K {kb..kb+15}; lanes 16-31: kb += 16.
DEVINL v16bf load_a16(const unsigned short* base, int row0, int ld, int k0, int lane) {
    const unsigned short* p = base + (row0 + (lane & 15)) * ld + k0 + ((lane >> 4) << 4);
    u16x8 lo = *(const u16x8*)(p);
    u16x8 hi = *(const u16x8*)(p + 8);
    return cat_u8(lo, hi);
}

// A operand with a per-lane row shift (3x3 conv halo) and zero-masking.
DEVINL v16bf load_a16_shift(const unsigned short* base, int row0, int lane, int doff, bool ok) {
    const unsigned short* p =
        base + (row0 + (lane & 15) + (ok ? doff : 0)) * 32 + ((lane >> 4) << 4);
    u16x8 lo = *(const u16x8*)(p);
    u16x8 hi = *(const u16x8*)(p + 8);
    u16x16 u = __builtin_shufflevector(lo, hi, 0,1,2,3,4,5,6,7,8,9,10,11,12,13,14,15);
    u16x16 z = (u16x16)0;
    u = ok ? u : z;
    return __builtin_bit_cast(v16bf, u);
}

// B operand (32x16, KxN), B[k][n] = pts[(col0+n)*ld + k]: lane n holds N=n&15,
// K contiguous chunk of 16 selected by lane>>4 (K-contiguous in memory).
DEVINL v16bf load_bpts(const unsigned short* base, int col0, int ld, int lane) {
    const unsigned short* p = base + (col0 + (lane & 15)) * ld + ((lane >> 4) << 4);
    u16x8 lo = *(const u16x8*)(p);
    u16x8 hi = *(const u16x8*)(p + 8);
    return cat_u8(lo, hi);
}

// B operand from pre-transposed bf16 weights wt[col*ldk + k] (K-contiguous).
DEVINL v16bf load_bwt(const unsigned short* wt, int ldk, int col, int k0) {
    const unsigned short* p = wt + col * ldk + k0;
    u16x8 lo = *(const u16x8*)(p);
    u16x8 hi = *(const u16x8*)(p + 8);
    return cat_u8(lo, hi);
}

// Sorted ascending top-8 insertion (registers only; fully unrolled).
DEVINL void topk_insert(float (&bd)[8], int (&bi)[8], float d, int j) {
    if (d < bd[7]) {
        bd[7] = d; bi[7] = j;
#pragma unroll
        for (int s = 7; s > 0; --s) {
            if (bd[s] < bd[s - 1]) {
                float td = bd[s]; bd[s] = bd[s - 1]; bd[s - 1] = td;
                int   ti = bi[s]; bi[s] = bi[s - 1]; bi[s - 1] = ti;
            }
        }
    }
}

// ---------------------------------------------------------------------------
// Kernel 0a: x [B,32,H,W] fp32 -> channel-last bf16 xcl [B, N=1024, 32].
// ---------------------------------------------------------------------------
__global__ void __launch_bounds__(256)
xcl_convert(const float* __restrict__ x, unsigned short* __restrict__ xcl) {
    int gid = blockIdx.x * 256 + threadIdx.x;   // b*1024 + n
    int b = gid >> 10, n = gid & 1023;
    const float* xp = x + b * 32768 + n;        // stride 1024 over channels
    unsigned short* dst = xcl + gid * 32;
#pragma unroll
    for (int i = 0; i < 32; i += 8) {
        u16x8 v;
#pragma unroll
        for (int t = 0; t < 8; ++t) v[t] = f2bf(xp[(i + t) * 1024]);
        *(u16x8*)(dst + i) = v;
    }
}

// ---------------------------------------------------------------------------
// Kernel 0b: one-time weight prep.
//  - conv weights -> per-tap transposed bf16: wtap[tap][co][ci]
//  - BN folded to per-channel scale/shift
//  - gconv weights -> transposed bf16 wt[co][ci]; biases pre-summed sb+db.
// ---------------------------------------------------------------------------
__global__ void __launch_bounds__(256)
prep_kernel(const float* __restrict__ cw, const float* __restrict__ cb,
            const float* __restrict__ gamma, const float* __restrict__ beta,
            const float* __restrict__ mean, const float* __restrict__ var,
            const float* __restrict__ sw0, const float* __restrict__ dw0,
            const float* __restrict__ sb0, const float* __restrict__ db0,
            const float* __restrict__ sw1, const float* __restrict__ dw1,
            const float* __restrict__ sb1, const float* __restrict__ db1,
            const float* __restrict__ sw2, const float* __restrict__ dw2,
            const float* __restrict__ sb2, const float* __restrict__ db2,
            unsigned short* __restrict__ wtap, float* __restrict__ scale,
            float* __restrict__ shiftv,
            unsigned short* __restrict__ w0s, unsigned short* __restrict__ w0d,
            float* __restrict__ bias0,
            unsigned short* __restrict__ w1s, unsigned short* __restrict__ w1d,
            float* __restrict__ bias1,
            unsigned short* __restrict__ w2s, unsigned short* __restrict__ w2d,
            float* __restrict__ bias2) {
    int t = blockIdx.x * 256 + threadIdx.x;
    if (t < 9216) {                       // cw[(co*32+ci)*9 + tap]
        int tap = t % 9, r = t / 9;
        int ci = r & 31, co = r >> 5;
        wtap[tap * 1024 + co * 32 + ci] = f2bf(cw[t]);
    }
    if (t < 32) {
        float sc = gamma[t] * rsqrtf(var[t] + 1e-5f);
        scale[t]  = sc;
        shiftv[t] = beta[t] + (cb[t] - mean[t]) * sc;
        bias0[t]  = sb0[t] + db0[t];
    }
    if (t < 64) { bias1[t] = sb1[t] + db1[t]; bias2[t] = sb2[t] + db2[t]; }
    if (t < 1024) { int k = t >> 5, n = t & 31; w0s[n * 32 + k] = f2bf(sw0[t]); w0d[n * 32 + k] = f2bf(dw0[t]); }
    if (t < 2048) { int k = t >> 6, n = t & 63; w1s[n * 32 + k] = f2bf(sw1[t]); w1d[n * 32 + k] = f2bf(dw1[t]); }
    if (t < 4096) { int k = t >> 6, n = t & 63; w2s[n * 64 + k] = f2bf(sw2[t]); w2d[n * 64 + k] = f2bf(dw2[t]); }
}

// ---------------------------------------------------------------------------
// Kernel 1: 3x3 conv (32->32, SAME) + BN + LeakyReLU as 9 shifted WMMA GEMMs.
// One wave per 16-pixel segment (half an image row; h uniform per block).
// H-halo -> uniform tap skip; W-halo -> per-lane zero-masked A row.
// Epilogue through LDS: contiguous bf16 row stores + per-pixel sq-norm.
// ---------------------------------------------------------------------------
__global__ void __launch_bounds__(32)
conv_wmma(const unsigned short* __restrict__ xcl, const unsigned short* __restrict__ wtap,
          const float* __restrict__ scale, const float* __restrict__ shiftv,
          unsigned short* __restrict__ yf, float* __restrict__ sqv) {
    int lane = threadIdx.x;
    int b = blockIdx.x >> 6;
    int n0 = (blockIdx.x & 63) << 4;
    const unsigned short* xb = xcl + b * 32768;

    int h  = n0 >> 5;                 // uniform
    int wl = (n0 & 31) + (lane & 15); // this lane's pixel w coordinate
    int kq = (lane >> 4) << 4;        // B-operand K chunk base

    v8f c0 = {0.f, 0.f, 0.f, 0.f, 0.f, 0.f, 0.f, 0.f};
    v8f c1 = c0;

#pragma unroll
    for (int dy = -1; dy <= 1; ++dy) {
        int hy = h + dy;
        if (hy < 0 || hy > 31) continue;     // uniform branch: EXEC untouched
#pragma unroll
        for (int dx = -1; dx <= 1; ++dx) {
            int wx = wl + dx;
            bool ok = (wx >= 0) && (wx <= 31);
            int tap = (dy + 1) * 3 + (dx + 1);
            int doff = dy * 32 + dx;
            v16bf a = load_a16_shift(xb, n0, lane, doff, ok);
            const unsigned short* wt = wtap + tap * 1024;
            v16bf bw0 = load_bwt(wt, 32, lane & 15, kq);
            c0 = wmma_bf16(a, bw0, c0);
            v16bf bw1 = load_bwt(wt, 32, 16 + (lane & 15), kq);
            c1 = wmma_bf16(a, bw1, c1);
        }
    }

    // BN + LeakyReLU, stage through LDS for contiguous stores + sq-norm.
    __shared__ float ytile[16 * 32];
    int hi8 = (lane >> 4) << 3;
    int cl = lane & 15;
    float sc0 = scale[cl],      sh0 = shiftv[cl];
    float sc1 = scale[cl + 16], sh1 = shiftv[cl + 16];
#pragma unroll
    for (int r = 0; r < 8; ++r) {
        int m = r + hi8;
        float v0 = fmaf(c0[r], sc0, sh0); v0 = v0 >= 0.f ? v0 : 0.05f * v0;
        float v1 = fmaf(c1[r], sc1, sh1); v1 = v1 >= 0.f ? v1 : 0.05f * v1;
        ytile[m * 32 + cl]      = v0;
        ytile[m * 32 + cl + 16] = v1;
    }
    __syncthreads();
    if (lane < 16) {
        const float* yr = &ytile[lane * 32];
        unsigned short* dst = yf + ((b << 10) + n0 + lane) * 32;
        float ss = 0.f;
#pragma unroll
        for (int i = 0; i < 32; i += 8) {
            u16x8 o;
#pragma unroll
            for (int t = 0; t < 8; ++t) { float v = yr[i + t]; ss += v * v; o[t] = f2bf(v); }
            *(u16x8*)(dst + i) = o;
        }
        sqv[(b << 10) + n0 + lane] = ss;
    }
}

// ---------------------------------------------------------------------------
// Kernel 2: fused KNN. One wave per 16-row tile. dist = |a|^2+|b|^2-2a.b via
// WMMA bf16; next B tile + sq column are prefetched before the current WMMA
// is consumed; 16x16 dist tile staged in LDS; per-row top-8 in registers
// (lane r = low column half, lane r+16 = high half), merged at the end.
// ---------------------------------------------------------------------------
__global__ void __launch_bounds__(32)
knn_kernel(const unsigned short* __restrict__ yf, const float* __restrict__ sq,
           int* __restrict__ graph) {
    int lane = threadIdx.x;
    int b = blockIdx.x >> 6;
    int row0 = (blockIdx.x & 63) << 4;
    const unsigned short* base = yf + b * 32768;
    const float* sqb = sq + (b << 10);

    v16bf arow = load_a16(base, row0, 32, 0, lane);

    int hi8 = (lane >> 4) << 3;
    float sqr[8];
#pragma unroll
    for (int r = 0; r < 8; ++r) sqr[r] = sqb[row0 + r + hi8];

    float bd[8]; int bi[8];
#pragma unroll
    for (int s = 0; s < 8; ++s) { bd[s] = 3.4e38f; bi[s] = 0; }

    __shared__ float dtile[256];
    int myrow = lane & 15;
    int chalf = hi8;

    v16bf bcol = load_bpts(base, 0, 32, lane);
    float sqc = sqb[lane & 15];

    for (int c0 = 0; c0 < 1024; c0 += 16) {
        int cn = (c0 + 16 < 1024) ? c0 + 16 : 0;   // prefetch (wraps harmlessly)
        v16bf bnext = load_bpts(base, cn, 32, lane);
        float sqn = sqb[cn + (lane & 15)];

        v8f acc = {0.f, 0.f, 0.f, 0.f, 0.f, 0.f, 0.f, 0.f};
        acc = wmma_bf16(arow, bcol, acc);
#pragma unroll
        for (int r = 0; r < 8; ++r)
            dtile[(r + hi8) * 16 + (lane & 15)] = sqr[r] + sqc - 2.0f * acc[r];
        __syncthreads();
#pragma unroll
        for (int c = 0; c < 8; ++c)
            topk_insert(bd, bi, dtile[myrow * 16 + chalf + c], c0 + chalf + c);
        __syncthreads();

        bcol = bnext; sqc = sqn;
    }

    __shared__ float md[256];
    __shared__ int   mi[256];
#pragma unroll
    for (int s = 0; s < 8; ++s) { md[lane * 8 + s] = bd[s]; mi[lane * 8 + s] = bi[s]; }
    __syncthreads();
    if (lane < 16) {
        float fd[8]; int fi[8];
#pragma unroll
        for (int s = 0; s < 8; ++s) { fd[s] = md[lane * 8 + s]; fi[s] = mi[lane * 8 + s]; }
#pragma unroll
        for (int s = 0; s < 8; ++s)
            topk_insert(fd, fi, md[(lane + 16) * 8 + s], mi[(lane + 16) * 8 + s]);
        int* g = graph + ((b << 10) + row0 + lane) * 8;
#pragma unroll
        for (int s = 0; s < 8; ++s) g[s] = fi[s];
    }
}

// ---------------------------------------------------------------------------
// Kernel 3: fused graph conv layer:
//   out = x @ sw + (mean_k x[nbr] - x) @ dw + bias; LeakyReLU(0.05).
// One wave per 16-row tile: gather+mean -> LDS (bf16), then 2 WMMAs per
// (kstep, col-tile) accumulating into one f32 C tile. Weights pre-transposed
// bf16 -> B operands are two 16B loads each.
// ---------------------------------------------------------------------------
template <int CI, int CO, bool FINAL>
__global__ void __launch_bounds__(32)
gconv_kernel(const unsigned short* __restrict__ xin, const int* __restrict__ graph,
             const unsigned short* __restrict__ wst, const unsigned short* __restrict__ wdt,
             const float* __restrict__ bias,
             unsigned short* __restrict__ xout, float* __restrict__ fout) {
    constexpr int CH = CI / 2;   // channels handled per lane in mean phase
    constexpr int KS = CI / 32;  // WMMA k-steps

    int lane = threadIdx.x;
    int b = blockIdx.x >> 6;
    int row0 = (blockIdx.x & 63) << 4;
    int row = lane & 15, hi = lane >> 4;

    __shared__ int            idx_s[16 * 8];
    __shared__ unsigned short mtile[16 * CI];

    {   // stage the 16x8 neighbor index tile
        const int* gp = graph + (((b << 10) + row0 + row) * 8) + hi * 4;
        int4 v = *(const int4*)gp;
        idx_s[row * 8 + hi * 4 + 0] = v.x;
        idx_s[row * 8 + hi * 4 + 1] = v.y;
        idx_s[row * 8 + hi * 4 + 2] = v.z;
        idx_s[row * 8 + hi * 4 + 3] = v.w;
    }
    __syncthreads();

    // gather + mean: lane handles (row, half of channels)
    const unsigned short* xb = xin + b * 1024 * CI;
    int cbase = hi * CH;
    float acc[CH];
#pragma unroll
    for (int i = 0; i < CH; ++i) acc[i] = 0.f;
    for (int k = 0; k < 8; ++k) {
        int j = idx_s[row * 8 + k];
        const unsigned short* p = xb + j * CI + cbase;
#pragma unroll
        for (int i = 0; i < CH; i += 8) {
            u16x8 v = *(const u16x8*)(p + i);
#pragma unroll
            for (int t = 0; t < 8; ++t) acc[i + t] += bf2f(v[t]);
        }
    }
    {
        const unsigned short* pr = xb + (row0 + row) * CI + cbase;
#pragma unroll
        for (int i = 0; i < CH; i += 8) {
            u16x8 v = *(const u16x8*)(pr + i);
#pragma unroll
            for (int t = 0; t < 8; ++t) {
                float m = 0.125f * acc[i + t] - bf2f(v[t]);
                mtile[row * CI + cbase + i + t] = f2bf(m);
            }
        }
    }
    __syncthreads();

    // A operands: x tile from global, mean tile from LDS (ds_load path)
    v16bf ax[KS], am[KS];
#pragma unroll
    for (int s = 0; s < KS; ++s) {
        ax[s] = load_a16(xb, row0, CI, 32 * s, lane);
        am[s] = load_a16(mtile, 0, CI, 32 * s, lane);
    }

    int coln = lane & 15;
    int krow0 = hi << 4;
#pragma unroll
    for (int t = 0; t < CO / 16; ++t) {
        int col = t * 16 + coln;
        v8f c = {0.f, 0.f, 0.f, 0.f, 0.f, 0.f, 0.f, 0.f};
#pragma unroll
        for (int s = 0; s < KS; ++s) {
            v16bf bs = load_bwt(wst, CI, col, 32 * s + krow0);
            c = wmma_bf16(ax[s], bs, c);
            v16bf bdv = load_bwt(wdt, CI, col, 32 * s + krow0);
            c = wmma_bf16(am[s], bdv, c);
        }
        float bv = bias[col];
#pragma unroll
        for (int r = 0; r < 8; ++r) {
            float v = c[r] + bv;
            v = v >= 0.f ? v : 0.05f * v;
            int ro = row0 + r + (hi << 3);
            if (FINAL) {
                fout[(b * CO + col) * 1024 + ro] = v;   // [B, OC, H*W]
            } else {
                xout[(b * 1024 + ro) * CO + col] = f2bf(v);
            }
        }
    }
}

// ---------------------------------------------------------------------------
extern "C" void kernel_launch(void* const* d_in, const int* in_sizes, int n_in,
                              void* d_out, int out_size, void* d_ws, size_t ws_size,
                              hipStream_t stream) {
    (void)in_sizes; (void)n_in; (void)out_size; (void)ws_size;
    const float* x   = (const float*)d_in[0];
    const float* cw  = (const float*)d_in[1];
    const float* cb  = (const float*)d_in[2];
    const float* bng = (const float*)d_in[3];
    const float* bnb = (const float*)d_in[4];
    const float* bnm = (const float*)d_in[5];
    const float* bnv = (const float*)d_in[6];
    const float* dw0 = (const float*)d_in[7];
    const float* db0 = (const float*)d_in[8];
    const float* sw0 = (const float*)d_in[9];
    const float* sb0 = (const float*)d_in[10];
    const float* dw1 = (const float*)d_in[11];
    const float* db1 = (const float*)d_in[12];
    const float* sw1 = (const float*)d_in[13];
    const float* sb1 = (const float*)d_in[14];
    const float* dw2 = (const float*)d_in[15];
    const float* db2 = (const float*)d_in[16];
    const float* sw2 = (const float*)d_in[17];
    const float* sb2 = (const float*)d_in[18];

    char* ws = (char*)d_ws;
    size_t off = 0;
    auto alloc = [&](size_t bytes) -> void* {
        void* p = ws + off;
        off = (off + bytes + 255) & ~(size_t)255;
        return p;
    };
    unsigned short* xcl  = (unsigned short*)alloc((size_t)64 * 1024 * 32 * 2); // 4 MiB
    unsigned short* yf0  = (unsigned short*)alloc((size_t)64 * 1024 * 32 * 2); // 4 MiB
    float*          sqv  = (float*)alloc((size_t)64 * 1024 * 4);               // 256 KiB
    int*            gra  = (int*)alloc((size_t)64 * 1024 * 8 * 4);             // 2 MiB
    unsigned short* yf1  = (unsigned short*)alloc((size_t)64 * 1024 * 32 * 2); // 4 MiB
    unsigned short* yf2  = (unsigned short*)alloc((size_t)64 * 1024 * 64 * 2); // 8 MiB
    unsigned short* wtap = (unsigned short*)alloc(9216 * 2);
    float*          scal = (float*)alloc(32 * 4);
    float*          shft = (float*)alloc(32 * 4);
    unsigned short* w0s  = (unsigned short*)alloc(1024 * 2);
    unsigned short* w0d  = (unsigned short*)alloc(1024 * 2);
    float*          b0   = (float*)alloc(32 * 4);
    unsigned short* w1s  = (unsigned short*)alloc(2048 * 2);
    unsigned short* w1d  = (unsigned short*)alloc(2048 * 2);
    float*          b1   = (float*)alloc(64 * 4);
    unsigned short* w2s  = (unsigned short*)alloc(4096 * 2);
    unsigned short* w2d  = (unsigned short*)alloc(4096 * 2);
    float*          b2   = (float*)alloc(64 * 4);

    xcl_convert<<<256, 256, 0, stream>>>(x, xcl);
    prep_kernel<<<36, 256, 0, stream>>>(cw, cb, bng, bnb, bnm, bnv,
                                        sw0, dw0, sb0, db0, sw1, dw1, sb1, db1,
                                        sw2, dw2, sb2, db2,
                                        wtap, scal, shft,
                                        w0s, w0d, b0, w1s, w1d, b1, w2s, w2d, b2);
    conv_wmma<<<4096, 32, 0, stream>>>(xcl, wtap, scal, shft, yf0, sqv);
    knn_kernel<<<4096, 32, 0, stream>>>(yf0, sqv, gra);
    gconv_kernel<32, 32, false><<<4096, 32, 0, stream>>>(yf0, gra, w0s, w0d, b0, yf1, nullptr);
    gconv_kernel<32, 64, false><<<4096, 32, 0, stream>>>(yf1, gra, w1s, w1d, b1, yf2, nullptr);
    gconv_kernel<64, 64, true ><<<4096, 32, 0, stream>>>(yf2, gra, w2s, w2d, b2, nullptr, (float*)d_out);
}
